// MiniMultiHeadAttention_3478923510323
// MI455X (gfx1250) — compile-verified, matrix operand order checked by reading, not run on hardware
//
#include <hip/hip_runtime.h>

// ---------------------------------------------------------------------------
// MiniMultiHeadAttention on MI455X (gfx1250, wave32, WMMA + TDM)
//   B=4 S=2048 H=1024 NH=16 HD=64
// Pipeline: fp32->bf16 convert | QKV GEMMs (WMMA bf16, TDM-staged weights) |
// flash-attention (WMMA bf16, TDM-staged K tiles, online softmax) |
// output GEMM (WMMA bf16 -> fp32 out)
// ---------------------------------------------------------------------------

typedef __bf16 bf16_t;
typedef __attribute__((ext_vector_type(16))) __bf16 v16bf;
typedef __attribute__((ext_vector_type(8)))  __bf16 v8bf;
typedef __attribute__((ext_vector_type(8)))  float  v8f;
typedef __attribute__((ext_vector_type(4)))  unsigned int v4u;
typedef __attribute__((ext_vector_type(8)))  int v8i;
typedef __attribute__((ext_vector_type(4)))  int v4i;

constexpr int kH  = 1024;
constexpr int kNH = 16;
constexpr int kHD = 64;
constexpr int kB  = 4;
constexpr int kS  = 2048;
constexpr int kBS = kB * kS; // 8192

__device__ __forceinline__ v16bf cat16(v8bf lo, v8bf hi) {
  return __builtin_shufflevector(lo, hi, 0,1,2,3,4,5,6,7,8,9,10,11,12,13,14,15);
}

__device__ __forceinline__ v8f wmma_bf16(v16bf a, v16bf b, v8f c) {
  // v_wmma_f32_16x16x32_bf16  D = A*B + C  (fp32 accumulate)
  return __builtin_amdgcn_wmma_f32_16x16x32_bf16(false, a, false, b, (short)0, c,
                                                 false, false);
}

// ---------------------------------------------------------------------------
// Tensor Data Mover: 2D tile (tile0 contiguous elems x tile1 rows, bf16,
// row stride = strideElems) from global memory into LDS at ldsAddr.
// Builds the D# per CDNA5 ISA (group0 + group1, 2D form).
// Issue from one wave; completion tracked with TENSORcnt.
// ---------------------------------------------------------------------------
__device__ __forceinline__ void tdm_load_2d(unsigned int ldsAddr,
                                            const void* gptr,
                                            unsigned int tile0,
                                            unsigned int tile1,
                                            unsigned int strideElems) {
  unsigned long long ga = (unsigned long long)(uintptr_t)gptr;
  v4u g0;
  g0[0] = 1u;                                  // count=1 (valid user descriptor)
  g0[1] = ldsAddr;                             // bits  63:32  lds_addr (bytes)
  g0[2] = (unsigned int)ga;                    // bits  95:64  global_addr lo
  g0[3] = (unsigned int)((ga >> 32) & 0x01FFFFFFull) | (2u << 30); // addr hi + type=2

  v8i g1;
  g1[0] = 1 << 16;                             // data_size = 1 (2-byte elements)
  g1[1] = (int)((tile0 & 0xFFFFu) << 16);      // tensor_dim0[15:0]  (== tile0)
  g1[2] = (int)(((tile0 >> 16) & 0xFFFFu) |    // tensor_dim0[31:16]
                ((tile1 & 0xFFFFu) << 16));    // tensor_dim1[15:0]  (== tile1)
  g1[3] = (int)(((tile1 >> 16) & 0xFFFFu) |    // tensor_dim1[31:16]
                ((tile0 & 0xFFFFu) << 16));    // tile_dim0
  g1[4] = (int)(tile1 & 0xFFFFu);              // tile_dim1 (tile_dim2 = 0)
  g1[5] = (int)strideElems;                    // tensor_dim0_stride[31:0]
  g1[6] = 0;                                   // stride[47:32], dim1_stride lo
  g1[7] = 0;

  v4i z4 = {0, 0, 0, 0};
#if defined(__clang_major__) && __clang_major__ >= 23
  v8i z8 = {0, 0, 0, 0, 0, 0, 0, 0};
  __builtin_amdgcn_tensor_load_to_lds(g0, g1, z4, z4, z8, 0);
#else
  __builtin_amdgcn_tensor_load_to_lds(g0, g1, z4, z4, 0);
#endif
}

// ---------------------------------------------------------------------------
// fp32 -> bf16 conversion (vectorized 4-wide)
// ---------------------------------------------------------------------------
__global__ void cvt_f32_bf16(const float* __restrict__ src,
                             bf16_t* __restrict__ dst, int n4) {
  int i = blockIdx.x * blockDim.x + threadIdx.x;
  if (i >= n4) return;
  float4 f = reinterpret_cast<const float4*>(src)[i];
  union { bf16_t h[4]; uint2 u; } o;
  o.h[0] = (bf16_t)f.x; o.h[1] = (bf16_t)f.y;
  o.h[2] = (bf16_t)f.z; o.h[3] = (bf16_t)f.w;
  reinterpret_cast<uint2*>(dst)[i] = o.u;
}

// ---------------------------------------------------------------------------
// C[M,N] = A[M,K](bf16) @ W[N,K]^T (bf16) + bias   (nn.Linear convention)
// Block = 256 threads = 8 waves; block tile 128(M) x 64(N); wave tile 16x64.
// Weight k-slice DMA'd into double-buffered LDS by the TDM (wave 0 drives it,
// TENSORcnt-pipelined one tile ahead); all 8 waves share it.
// ---------------------------------------------------------------------------
template<bool OUT_F32>
__global__ __launch_bounds__(256)
void linear_wmma(const bf16_t* __restrict__ A, const bf16_t* __restrict__ W,
                 const float* __restrict__ bias,
                 bf16_t* __restrict__ outB, float* __restrict__ outF,
                 int M, int N, int K) {
  __shared__ bf16_t ldsW[2][64][32]; // double-buffered 64 N-rows x 32 K (8 KB)

  const int tid   = threadIdx.x;
  const int lane  = tid & 31;
  const int wave  = tid >> 5;
  const int row   = lane & 15;   // M (A-frag) or N (B-frag/C) index mod 16
  const int hi    = lane >> 4;   // high half-wave
  const int kLane = hi * 8;      // per-ISA 16-bit A layout: hi lanes start K=8
  const int mBase = blockIdx.y * 128 + wave * 16;
  const int nBase = blockIdx.x * 64;

  v8f acc[4] = {};

  const bf16_t* aRow  = A + (size_t)(mBase + row) * K;
  const bf16_t* wBase = W + (size_t)nBase * K;
  const unsigned int ldsWAddr[2] = {
      (unsigned int)(uintptr_t)&ldsW[0][0][0],
      (unsigned int)(uintptr_t)&ldsW[1][0][0]};

  const int nIter = K / 32;
  if (wave == 0) tdm_load_2d(ldsWAddr[0], wBase, 32, 64, (unsigned int)K);

  for (int it = 0; it < nIter; ++it) {
    const int cur = it & 1;
    const int k0  = it * 32;
    __syncthreads(); // prev iteration's reads of ldsW[cur^1] complete
    if (wave == 0) {
      if (it + 1 < nIter) {
        tdm_load_2d(ldsWAddr[cur ^ 1], wBase + (it + 1) * 32, 32, 64,
                    (unsigned int)K);
        __builtin_amdgcn_s_wait_tensorcnt(1); // tile `it` landed; it+1 in flight
      } else {
        __builtin_amdgcn_s_wait_tensorcnt(0);
      }
    }
    __syncthreads(); // ldsW[cur] visible to all waves

    v8bf alo = *reinterpret_cast<const v8bf*>(aRow + k0 + kLane);
    v8bf ahi = *reinterpret_cast<const v8bf*>(aRow + k0 + kLane + 16);
    v16bf afrag = cat16(alo, ahi);

#pragma unroll
    for (int nt = 0; nt < 4; ++nt) {
      v8bf blo = *reinterpret_cast<const v8bf*>(&ldsW[cur][nt * 16 + row][kLane]);
      v8bf bhi = *reinterpret_cast<const v8bf*>(&ldsW[cur][nt * 16 + row][kLane + 16]);
      acc[nt] = wmma_bf16(afrag, cat16(blo, bhi), acc[nt]);
    }
  }

#pragma unroll
  for (int nt = 0; nt < 4; ++nt) {
    int col = nBase + nt * 16 + row;
    float bv = bias[col];
#pragma unroll
    for (int r = 0; r < 8; ++r) {
      int orow = mBase + r + hi * 8; // C layout: VGPR r -> M=r (lo) / r+8 (hi)
      float v = acc[nt][r] + bv;
      if (OUT_F32) outF[(size_t)orow * N + col] = v;
      else         outB[(size_t)orow * N + col] = (bf16_t)v;
    }
  }
}

// ---------------------------------------------------------------------------
// Flash attention: one block = (b,h) x 128 query rows; 8 waves x 16 q rows.
// 32-key tiles: K DMA'd row-major into double-buffered LDS by the TDM
// (one tile ahead, TENSORcnt-pipelined); V staged transposed in LDS manually.
// scores = Q K^T / 8 ; online softmax ; O += P V.  All matmuls via WMMA.
// ---------------------------------------------------------------------------
__global__ __launch_bounds__(256)
void attn_wmma(const bf16_t* __restrict__ Q, const bf16_t* __restrict__ Kt,
               const bf16_t* __restrict__ V, bf16_t* __restrict__ O) {
  __shared__ bf16_t ldsK [2][32][64];   // 8 KB  key-tile x2, row-major [key][d]
  __shared__ bf16_t ldsVt[64][32];      // 4 KB  value-tile transposed [d][key]
  __shared__ bf16_t ldsP [8][16][32];   // 8 KB  per-wave P transpose scratch

  const int tid   = threadIdx.x;
  const int lane  = tid & 31;
  const int wave  = tid >> 5;
  const int row   = lane & 15;
  const int hi    = lane >> 4;
  const int kLane = hi * 8;
  const int b     = blockIdx.y >> 4;
  const int h     = blockIdx.y & 15;
  const int qBase = blockIdx.x * 128 + wave * 16;

  // Q fragments (d = 0..63 -> two 16x32 A-fragments), loaded once
  const size_t rowQ = (size_t)(b * kS + qBase + row) * kH + h * kHD;
  v16bf qf[2];
#pragma unroll
  for (int kh = 0; kh < 2; ++kh) {
    v8bf lo  = *reinterpret_cast<const v8bf*>(Q + rowQ + kh * 32 + kLane);
    v8bf hi8 = *reinterpret_cast<const v8bf*>(Q + rowQ + kh * 32 + kLane + 16);
    qf[kh] = cat16(lo, hi8);
  }

  v8f oacc[4] = {};
  float m[8], l[8];
#pragma unroll
  for (int r = 0; r < 8; ++r) { m[r] = -INFINITY; l[r] = 0.0f; }

  const size_t kvBase = (size_t)(b * kS) * kH + h * kHD;
  const unsigned int ldsKAddr[2] = {
      (unsigned int)(uintptr_t)&ldsK[0][0][0],
      (unsigned int)(uintptr_t)&ldsK[1][0][0]};

  const int nIter = kS / 32; // 64
  if (wave == 0) tdm_load_2d(ldsKAddr[0], Kt + kvBase, 64, 32, kH);

  for (int it = 0; it < nIter; ++it) {
    const int cur = it & 1;
    const int kt  = it * 32;
    __syncthreads(); // prior iteration's LDS reads complete
    { // cooperative V tile load + transpose: 32 keys x 64 d, 16B/thread
      int r = tid >> 3;
      int c = (tid & 7) * 8;
      size_t g = kvBase + (size_t)(kt + r) * kH + c;
      v8bf vv = *reinterpret_cast<const v8bf*>(V + g);
#pragma unroll
      for (int j = 0; j < 8; ++j) ldsVt[c + j][r] = vv[j]; // transpose store
      if (kt + 32 < kS)
        __builtin_prefetch(V + g + (size_t)32 * kH, 0, 1);
    }
    if (wave == 0) { // TDM: prefetch next K tile, ensure current tile landed
      if (it + 1 < nIter) {
        tdm_load_2d(ldsKAddr[cur ^ 1], Kt + kvBase + (size_t)(kt + 32) * kH,
                    64, 32, kH);
        __builtin_amdgcn_s_wait_tensorcnt(1);
      } else {
        __builtin_amdgcn_s_wait_tensorcnt(0);
      }
    }
    __syncthreads();

    // ---- scores: two 16(q) x 16(key) tiles, each accumulated over d=64 ----
    v8f sacc[2] = {};
#pragma unroll
    for (int st = 0; st < 2; ++st) {
#pragma unroll
      for (int kh = 0; kh < 2; ++kh) {
        v8bf blo = *reinterpret_cast<const v8bf*>(
            &ldsK[cur][st * 16 + row][kh * 32 + kLane]);
        v8bf bhi = *reinterpret_cast<const v8bf*>(
            &ldsK[cur][st * 16 + row][kh * 32 + kLane + 16]);
        sacc[st] = wmma_bf16(qf[kh], cat16(blo, bhi), sacc[st]);
      }
    }

    // ---- online softmax over this 32-key tile ----
    float tmax[8], p0[8], p1[8], psum[8];
#pragma unroll
    for (int r = 0; r < 8; ++r) {
      float s0 = sacc[0][r] * 0.125f;  // 1/sqrt(64)
      float s1 = sacc[1][r] * 0.125f;
      sacc[0][r] = s0; sacc[1][r] = s1;
      tmax[r] = fmaxf(s0, s1);
    }
    for (int off = 1; off < 16; off <<= 1) { // reduce within 16-lane group
#pragma unroll
      for (int r = 0; r < 8; ++r)
        tmax[r] = fmaxf(tmax[r], __shfl_xor(tmax[r], off, 32));
    }
#pragma unroll
    for (int r = 0; r < 8; ++r) {
      float mn    = fmaxf(m[r], tmax[r]);
      float alpha = __expf(m[r] - mn);
      m[r] = mn;
      p0[r] = __expf(sacc[0][r] - mn);
      p1[r] = __expf(sacc[1][r] - mn);
      psum[r] = p0[r] + p1[r];
      l[r] *= alpha;
#pragma unroll
      for (int nt = 0; nt < 4; ++nt) oacc[nt][r] *= alpha;
    }
    for (int off = 1; off < 16; off <<= 1) {
#pragma unroll
      for (int r = 0; r < 8; ++r)
        psum[r] += __shfl_xor(psum[r], off, 32);
    }
#pragma unroll
    for (int r = 0; r < 8; ++r) l[r] += psum[r];

    // ---- transpose P (C-layout -> A-fragment layout) through per-wave LDS ---
#pragma unroll
    for (int r = 0; r < 8; ++r) {
      ldsP[wave][r + hi * 8][row]      = (bf16_t)p0[r];
      ldsP[wave][r + hi * 8][row + 16] = (bf16_t)p1[r];
    }
    v8bf plo = *reinterpret_cast<const v8bf*>(&ldsP[wave][row][kLane]);
    v8bf phi = *reinterpret_cast<const v8bf*>(&ldsP[wave][row][kLane + 16]);
    v16bf pf = cat16(plo, phi);

    // ---- O += P(16x32) * V(32x64): B-frags contiguous from transposed LDS --
#pragma unroll
    for (int nt = 0; nt < 4; ++nt) {
      v8bf blo = *reinterpret_cast<const v8bf*>(&ldsVt[nt * 16 + row][kLane]);
      v8bf bhi = *reinterpret_cast<const v8bf*>(&ldsVt[nt * 16 + row][kLane + 16]);
      oacc[nt] = wmma_bf16(pf, cat16(blo, bhi), oacc[nt]);
    }
  }

  // ---- epilogue: normalize and store O (bf16, [b,s,h,d] flattened [BS][H]) --
#pragma unroll
  for (int r = 0; r < 8; ++r) {
    float inv = 1.0f / l[r];
    int orow = b * kS + qBase + r + hi * 8;
#pragma unroll
    for (int nt = 0; nt < 4; ++nt) {
      int col = h * kHD + nt * 16 + row;
      O[(size_t)orow * kH + col] = (bf16_t)(oacc[nt][r] * inv);
    }
  }
}

// ---------------------------------------------------------------------------
// Host launcher
// ---------------------------------------------------------------------------
extern "C" void kernel_launch(void* const* d_in, const int* in_sizes, int n_in,
                              void* d_out, int out_size, void* d_ws, size_t ws_size,
                              hipStream_t stream) {
  const float* hs = (const float*)d_in[0];
  const float* wq = (const float*)d_in[1];
  const float* bq = (const float*)d_in[2];
  const float* wk = (const float*)d_in[3];
  const float* bk = (const float*)d_in[4];
  const float* wv = (const float*)d_in[5];
  const float* bv = (const float*)d_in[6];
  const float* wo = (const float*)d_in[7];
  const float* bo = (const float*)d_in[8];

  const size_t nX = (size_t)kBS * kH; // 8,388,608
  const size_t nW = (size_t)kH * kH;  // 1,048,576

  bf16_t* Xb  = (bf16_t*)d_ws;
  bf16_t* Qb  = Xb + nX;
  bf16_t* Kb  = Qb + nX;
  bf16_t* Vb  = Kb + nX;
  bf16_t* Ob  = Vb + nX;
  bf16_t* Wqb = Ob + nX;
  bf16_t* Wkb = Wqb + nW;
  bf16_t* Wvb = Wkb + nW;
  bf16_t* Wob = Wvb + nW;

  // fp32 -> bf16 (X + 4 weight matrices)
  cvt_f32_bf16<<<(int)(nX / 4 / 256), 256, 0, stream>>>(hs, Xb, (int)(nX / 4));
  cvt_f32_bf16<<<(int)(nW / 4 / 256), 256, 0, stream>>>(wq, Wqb, (int)(nW / 4));
  cvt_f32_bf16<<<(int)(nW / 4 / 256), 256, 0, stream>>>(wk, Wkb, (int)(nW / 4));
  cvt_f32_bf16<<<(int)(nW / 4 / 256), 256, 0, stream>>>(wv, Wvb, (int)(nW / 4));
  cvt_f32_bf16<<<(int)(nW / 4 / 256), 256, 0, stream>>>(wo, Wob, (int)(nW / 4));

  dim3 gGemm(kH / 64, kBS / 128); // (16, 64)
  linear_wmma<false><<<gGemm, 256, 0, stream>>>(Xb, Wqb, bq, Qb, nullptr, kBS, kH, kH);
  linear_wmma<false><<<gGemm, 256, 0, stream>>>(Xb, Wkb, bk, Kb, nullptr, kBS, kH, kH);
  linear_wmma<false><<<gGemm, 256, 0, stream>>>(Xb, Wvb, bv, Vb, nullptr, kBS, kH, kH);

  dim3 gAttn(kS / 128, kB * kNH);  // (16, 64)
  attn_wmma<<<gAttn, 256, 0, stream>>>(Qb, Kb, Vb, Ob);

  linear_wmma<true><<<gGemm, 256, 0, stream>>>(Ob, Wob, bo, nullptr, (float*)d_out,
                                               kBS, kH, kH);
}
